// MLA_70480413327859
// MI455X (gfx1250) — matrix operand order repaired and checked by reading, not compiled, not run on hardware
//
#include <hip/hip_runtime.h>
#include <hip/hip_bf16.h>

// ---------------------------------------------------------------------------
// MLA prefill for gfx1250 (MI455X, wave32, WMMA).
// Strategy: bf16 everywhere through v_wmma_f32_16x16x32_bf16 (fp32 accum).
// Entire bf16 working set (~31MB weights) is L2-resident (192MB), so GEMM
// fragments are loaded directly from global as 16B chunks.
// GEMM register blocking: 32x64 per wave (2x4 WMMA frags) -> 8 WMMAs per
// 6 fragment loads per k-step, ~2x better compute/load ratio than 2x2.
// ---------------------------------------------------------------------------

#define B_SZ     2
#define S_LEN    2048
#define D_MODEL  2048
#define H_NUM    16
#define DCQ      1536
#define DCKV     512
#define DROPE    64
#define QHD      192      // 128 nope + 64 rope
#define DV       128
#define BS       (B_SZ * S_LEN)   // 4096 rows

typedef __bf16 bf16;
typedef __attribute__((ext_vector_type(16))) __bf16 v16bf;
typedef __attribute__((ext_vector_type(8)))  __bf16 v8bf;
typedef __attribute__((ext_vector_type(8)))  float  v8f;

union AB16 { v16bf v; v8bf h[2]; };

// --- WMMA helper ------------------------------------------------------------
__device__ __forceinline__ v8f wmma_bf16(v16bf a, v16bf b, v8f c) {
  // (neg_a, A, neg_b, B, c_mod, C, reuse_a, reuse_b)
  return __builtin_amdgcn_wmma_f32_16x16x32_bf16(false, a, false, b,
                                                 (short)0, c, false, false);
}

// A fragment (16x32, M x K) from row-major matrix, leading dim ld.
// Layout: lanes 0-15 row M=lane hold K={0..7,16..23}; lanes 16-31 hold K={8..15,24..31}.
__device__ __forceinline__ v16bf load_frag_a(const bf16* __restrict__ base,
                                             int ld, int row0, int k0) {
  int lane = threadIdx.x & 31;
  const bf16* src = base + (size_t)(row0 + (lane & 15)) * ld + k0 + ((lane >> 4) << 3);
  AB16 f;
  f.h[0] = *reinterpret_cast<const v8bf*>(src);
  f.h[1] = *reinterpret_cast<const v8bf*>(src + 16);
  return f.v;
}

// B fragment (32x16, K x N) where the N-major source is row-major W (N rows, K cols),
// i.e. B = W^T.  Layout: lanes 0-15 (N=lane) hold K={0..7,16..23}; lanes 16-31 hold K={8..15,24..31}.
__device__ __forceinline__ v16bf load_frag_b(const bf16* __restrict__ base,
                                             int ld, int n0, int k0) {
  int lane = threadIdx.x & 31;
  const bf16* src = base + (size_t)(n0 + (lane & 15)) * ld + k0 + ((lane >> 4) << 3);
  AB16 f;
  f.h[0] = *reinterpret_cast<const v8bf*>(src);
  f.h[1] = *reinterpret_cast<const v8bf*>(src + 16);
  return f.v;
}

// --- fp32 -> bf16 convert ---------------------------------------------------
__global__ void cvt_f32_bf16_kernel(const float* __restrict__ in,
                                    bf16* __restrict__ out, int n) {
  int i = blockIdx.x * blockDim.x + threadIdx.x;
  int stride = gridDim.x * blockDim.x;
  for (; i < n; i += stride) out[i] = (bf16)in[i];
}

// --- generic GEMM: C[f32, M x N] = A[bf16, M x K] * W[bf16, N x K]^T --------
// Block: 256 thr (8 waves). Tile 256(M) x 64(N); wave = 32x64 = 2x4 WMMA frags.
// Per k-step/wave: 6 fragment loads feed 8 WMMAs (vs 8 loads / 4 WMMAs at 2x2).
__global__ __launch_bounds__(256) void gemm_bf16_kernel(
    const bf16* __restrict__ A, int lda,
    const bf16* __restrict__ W, int ldw,
    float* __restrict__ C, int ldc, int K) {
  int wave = threadIdx.x >> 5;
  int lane = threadIdx.x & 31;
  int m0 = blockIdx.y * 256 + wave * 32;
  int n0 = blockIdx.x * 64;

  v8f acc[2][4] = {{{}, {}, {}, {}}, {{}, {}, {}, {}}};
  for (int k = 0; k < K; k += 32) {
    v16bf a0 = load_frag_a(A, lda, m0,      k);
    v16bf a1 = load_frag_a(A, lda, m0 + 16, k);
    v16bf b0 = load_frag_b(W, ldw, n0,      k);
    v16bf b1 = load_frag_b(W, ldw, n0 + 16, k);
    v16bf b2 = load_frag_b(W, ldw, n0 + 32, k);
    v16bf b3 = load_frag_b(W, ldw, n0 + 48, k);
    acc[0][0] = wmma_bf16(a0, b0, acc[0][0]);
    acc[0][1] = wmma_bf16(a0, b1, acc[0][1]);
    acc[0][2] = wmma_bf16(a0, b2, acc[0][2]);
    acc[0][3] = wmma_bf16(a0, b3, acc[0][3]);
    acc[1][0] = wmma_bf16(a1, b0, acc[1][0]);
    acc[1][1] = wmma_bf16(a1, b1, acc[1][1]);
    acc[1][2] = wmma_bf16(a1, b2, acc[1][2]);
    acc[1][3] = wmma_bf16(a1, b3, acc[1][3]);
  }
  int ln = lane & 15, hi = lane >> 4;
#pragma unroll
  for (int i = 0; i < 2; ++i)
#pragma unroll
    for (int j = 0; j < 4; ++j)
#pragma unroll
      for (int v = 0; v < 8; ++v)
        C[(size_t)(m0 + 16 * i + v + 8 * hi) * ldc + n0 + 16 * j + ln] = acc[i][j][v];
}

// --- RMS norm: one block per row; out = w * x * rsqrt(mean(x^2)+eps) (bf16) -
__global__ __launch_bounds__(256) void rmsnorm_kernel(
    const float* __restrict__ in, int ld_in, int ncols,
    const float* __restrict__ w, bf16* __restrict__ out, int ld_out) {
  int row = blockIdx.x;
  const float* x = in + (size_t)row * ld_in;
  __shared__ float red[256];
  float s = 0.f;
  for (int i = threadIdx.x; i < ncols; i += 256) { float v = x[i]; s += v * v; }
  red[threadIdx.x] = s;
  __syncthreads();
  for (int off = 128; off > 0; off >>= 1) {
    if ((int)threadIdx.x < off) red[threadIdx.x] += red[threadIdx.x + off];
    __syncthreads();
  }
  float r = rsqrtf(red[0] / (float)ncols + 1e-7f);
  for (int i = threadIdx.x; i < ncols; i += 256)
    out[(size_t)row * ld_out + i] = (bf16)(w[i] * x[i] * r);
}

// --- extract fp32 k_rope tail from ckv (ld 576, offset 512) -----------------
__global__ void extract_krope_kernel(const float* __restrict__ ckv,
                                     float* __restrict__ krope) {
  int i = blockIdx.x * 256 + threadIdx.x;
  if (i < BS * DROPE) {
    int r = i >> 6, j = i & 63;
    krope[i] = ckv[(size_t)r * 576 + 512 + j];
  }
}

// --- RoPE helper: reference does reshape(d/2,2).swap -> concat rotate -------
// out[j]    = x[2j]*cos_j   - x[2j+1]*sin_j    (j in 0..31)
// out[j+32] = x[2j+1]*cos_j + x[2j]*sin_j
__device__ __forceinline__ void rope_pair(const float* __restrict__ src, int j,
                                          float pos, float& lo, float& hi) {
  float x0 = src[2 * j], x1 = src[2 * j + 1];
  // inv_freq[j] = 10000^(-2j/64)
  float freq = __expf(-(float)(2 * j) * (1.0f / 64.0f) * 9.2103403719761836f);
  float ang = pos * freq;
  float c = __cosf(ang), s = __sinf(ang);
  lo = x0 * c - x1 * s;
  hi = x1 * c + x0 * s;
}

// --- q_full (BS x 3072 f32) -> query (b,h,s,192) bf16 with RoPE on last 64 --
__global__ __launch_bounds__(256) void rope_q_kernel(
    const float* __restrict__ qfull, const int* __restrict__ pos_ids,
    bf16* __restrict__ query) {
  int row = blockIdx.x;                 // b*S + s
  int b = row / S_LEN, sidx = row % S_LEN;
  float pos = (float)pos_ids[row];
  const float* q = qfull + (size_t)row * (H_NUM * QHD);
  for (int idx = threadIdx.x; idx < H_NUM * QHD; idx += 256) {
    int h = idx / QHD, d = idx % QHD;
    float val;
    if (d < 128) {
      val = q[h * QHD + d];
    } else {
      int j = d - 128;
      const float* src = q + h * QHD + 128;
      float lo, hi;
      if (j < 32) { rope_pair(src, j, pos, lo, hi); val = lo; }
      else        { rope_pair(src, j - 32, pos, lo, hi); val = hi; }
    }
    query[(((size_t)b * H_NUM + h) * S_LEN + sidx) * QHD + d] = (bf16)val;
  }
}

// --- kv_full (BS x 4096 f32) + krope -> key (b,h,s,192), vT (b,h,128,s) -----
__global__ __launch_bounds__(256) void kv_split_kernel(
    const float* __restrict__ kvfull, const float* __restrict__ krope,
    const int* __restrict__ pos_ids,
    bf16* __restrict__ key, bf16* __restrict__ vT) {
  int row = blockIdx.x;                 // b*S + s
  int b = row / S_LEN, sidx = row % S_LEN;
  float pos = (float)pos_ids[row];
  const float* kv = kvfull + (size_t)row * (H_NUM * 256);
  const float* kr = krope + (size_t)row * DROPE;
  for (int idx = threadIdx.x; idx < H_NUM * 256; idx += 256) {
    int h = idx >> 8, d = idx & 255;
    if (d < 128) {
      key[(((size_t)b * H_NUM + h) * S_LEN + sidx) * QHD + d] = (bf16)kv[h * 256 + d];
    } else {
      int dv = d - 128;   // value dim, stored transposed: (b,h,dv,s)
      vT[(((size_t)b * H_NUM + h) * DV + dv) * S_LEN + sidx] = (bf16)kv[h * 256 + 128 + dv];
    }
  }
  for (int idx = threadIdx.x; idx < H_NUM * DROPE; idx += 256) {
    int h = idx >> 6, j = idx & 63;
    float lo, hi, val;
    if (j < 32) { rope_pair(kr, j, pos, lo, hi); val = lo; }
    else        { rope_pair(kr, j - 32, pos, lo, hi); val = hi; }
    key[(((size_t)b * H_NUM + h) * S_LEN + sidx) * QHD + 128 + j] = (bf16)val;
  }
}

// --- flash attention: grid (S/128, H, B); 8 waves, wave owns 16 q-rows ------
__global__ __launch_bounds__(256) void flash_attn_kernel(
    const bf16* __restrict__ query, const bf16* __restrict__ key,
    const bf16* __restrict__ vT, bf16* __restrict__ attn) {
  __shared__ bf16 lds_p[8][16 * 32];    // per-wave P tile (16 rows x 32 keys)
  const float SCALE = 0.07216878364870323f;   // 1/sqrt(192)

  int wave = threadIdx.x >> 5;
  int lane = threadIdx.x & 31;
  int hi = lane >> 4, ln = lane & 15;
  int b = blockIdx.z, h = blockIdx.y;
  int qrow0 = blockIdx.x * 128 + wave * 16;

  const bf16* Q = query + (((size_t)b * H_NUM + h) * S_LEN) * QHD;
  const bf16* Km = key  + (((size_t)b * H_NUM + h) * S_LEN) * QHD;
  const bf16* V = vT    + (((size_t)b * H_NUM + h) * DV) * S_LEN;

  // preload Q fragments covering d = 0..191
  v16bf aq[6];
#pragma unroll
  for (int t = 0; t < 6; ++t) aq[t] = load_frag_a(Q, QHD, qrow0, 32 * t);

  v8f o[8];
#pragma unroll
  for (int j = 0; j < 8; ++j) o[j] = (v8f){};
  float mrun[8], lrun[8];
#pragma unroll
  for (int v = 0; v < 8; ++v) { mrun[v] = -1e30f; lrun[v] = 0.f; }

  for (int kt = 0; kt <= qrow0 + 15; kt += 32) {
    // scores: two 16x16 C-frags over 32 keys, K-dim = 192
    v8f s0 = (v8f){}, s1 = (v8f){};
#pragma unroll
    for (int t = 0; t < 6; ++t) {
      v16bf b0 = load_frag_b(Km, QHD, kt,      32 * t);
      v16bf b1 = load_frag_b(Km, QHD, kt + 16, 32 * t);
      s0 = wmma_bf16(aq[t], b0, s0);
      s1 = wmma_bf16(aq[t], b1, s1);
    }
    // online softmax over the 32-key slab
    float p0[8], p1[8], alpha[8];
#pragma unroll
    for (int v = 0; v < 8; ++v) {
      int q_abs = qrow0 + v + 8 * hi;
      int k0 = kt + ln, k1 = kt + 16 + ln;
      float x0 = (k0 <= q_abs) ? s0[v] * SCALE : -1e30f;
      float x1 = (k1 <= q_abs) ? s1[v] * SCALE : -1e30f;
      float cm = fmaxf(x0, x1);
      cm = fmaxf(cm, __shfl_xor(cm, 1, 32));
      cm = fmaxf(cm, __shfl_xor(cm, 2, 32));
      cm = fmaxf(cm, __shfl_xor(cm, 4, 32));
      cm = fmaxf(cm, __shfl_xor(cm, 8, 32));
      float newm = fmaxf(mrun[v], cm);
      alpha[v] = __expf(mrun[v] - newm);
      mrun[v] = newm;
      p0[v] = __expf(x0 - newm);
      p1[v] = __expf(x1 - newm);
      float ls = p0[v] + p1[v];
      ls += __shfl_xor(ls, 1, 32);
      ls += __shfl_xor(ls, 2, 32);
      ls += __shfl_xor(ls, 4, 32);
      ls += __shfl_xor(ls, 8, 32);
      lrun[v] = lrun[v] * alpha[v] + ls;
    }
    // rescale running output
#pragma unroll
    for (int j = 0; j < 8; ++j)
#pragma unroll
      for (int v = 0; v < 8; ++v) o[j][v] *= alpha[v];

    // C-layout P -> LDS (row-major 16x32) -> A-fragment layout (wave-local;
    // LDS ops are in-order within a wave, compiler inserts dscnt waits)
    bf16* P = lds_p[wave];
#pragma unroll
    for (int v = 0; v < 8; ++v) {
      int m = v + 8 * hi;
      P[m * 32 + ln]      = (bf16)p0[v];
      P[m * 32 + 16 + ln] = (bf16)p1[v];
    }
    v16bf ap;
    {
      int r = ln, kb = hi << 3;
#pragma unroll
      for (int i = 0; i < 8; ++i) ap[i]     = P[r * 32 + kb + i];
#pragma unroll
      for (int i = 0; i < 8; ++i) ap[8 + i] = P[r * 32 + kb + 16 + i];
    }
    // P @ V : V stored transposed (dims x seq), so B-frag is a row load
#pragma unroll
    for (int j = 0; j < 8; ++j) {
      v16bf bv = load_frag_b(V, S_LEN, 16 * j, kt);
      o[j] = wmma_bf16(ap, bv, o[j]);
    }
  }

  // normalize and store to attn (b, s, h*128 + d) bf16
#pragma unroll
  for (int j = 0; j < 8; ++j)
#pragma unroll
    for (int v = 0; v < 8; ++v) {
      int q_abs = qrow0 + v + 8 * hi;
      attn[((size_t)b * S_LEN + q_abs) * (H_NUM * DV) + h * DV + 16 * j + ln] =
          (bf16)(o[j][v] / lrun[v]);
    }
}

// ---------------------------------------------------------------------------
extern "C" void kernel_launch(void* const* d_in, const int* in_sizes, int n_in,
                              void* d_out, int out_size, void* d_ws, size_t ws_size,
                              hipStream_t stream) {
  const float* x        = (const float*)d_in[0];
  const int*   pos_ids  = (const int*)d_in[1];
  const float* w_q_down = (const float*)d_in[2];
  const float* q_norm_w = (const float*)d_in[3];
  const float* w_q_up   = (const float*)d_in[4];
  const float* w_kv_down= (const float*)d_in[5];
  const float* kv_norm_w= (const float*)d_in[6];
  const float* w_kv_up  = (const float*)d_in[7];
  const float* w_out    = (const float*)d_in[8];
  float* out = (float*)d_out;

  // bump allocator over d_ws (256B aligned)
  char* ws = (char*)d_ws;
  size_t off = 0;
  auto alloc = [&](size_t bytes) -> void* {
    void* p = ws + off;
    off += (bytes + 255) & ~(size_t)255;
    return p;
  };

  bf16* xb    = (bf16*)alloc((size_t)BS * D_MODEL * 2);
  bf16* wqd   = (bf16*)alloc((size_t)DCQ * D_MODEL * 2);
  bf16* wqu   = (bf16*)alloc((size_t)(H_NUM * QHD) * DCQ * 2);
  bf16* wkvd  = (bf16*)alloc((size_t)576 * D_MODEL * 2);
  bf16* wkvu  = (bf16*)alloc((size_t)(H_NUM * 256) * DCKV * 2);
  bf16* woutb = (bf16*)alloc((size_t)D_MODEL * (H_NUM * DV) * 2);
  bf16* q_c   = (bf16*)alloc((size_t)BS * DCQ * 2);
  bf16* ckv_c = (bf16*)alloc((size_t)BS * DCKV * 2);
  float* krope= (float*)alloc((size_t)BS * DROPE * 4);
  bf16* query = (bf16*)alloc((size_t)B_SZ * H_NUM * S_LEN * QHD * 2);
  bf16* keyb  = (bf16*)alloc((size_t)B_SZ * H_NUM * S_LEN * QHD * 2);
  bf16* vT    = (bf16*)alloc((size_t)B_SZ * H_NUM * DV * S_LEN * 2);
  bf16* attn  = (bf16*)alloc((size_t)BS * (H_NUM * DV) * 2);
  float* tmpA = (float*)alloc((size_t)BS * (H_NUM * QHD) * 4);   // 4096x3072
  float* tmpB = (float*)alloc((size_t)BS * (H_NUM * 256) * 4);   // 4096x4096

  // 1) fp32 -> bf16 conversions
  auto cvt = [&](const float* src, bf16* dst, int n) {
    cvt_f32_bf16_kernel<<<1024, 256, 0, stream>>>(src, dst, n);
  };
  cvt(x,         xb,    BS * D_MODEL);
  cvt(w_q_down,  wqd,   DCQ * D_MODEL);
  cvt(w_q_up,    wqu,   H_NUM * QHD * DCQ);
  cvt(w_kv_down, wkvd,  576 * D_MODEL);
  cvt(w_kv_up,   wkvu,  H_NUM * 256 * DCKV);
  cvt(w_out,     woutb, D_MODEL * H_NUM * DV);

  auto gemm = [&](const bf16* A, int lda, const bf16* W, int ldw,
                  float* C, int ldc, int M, int N, int K) {
    dim3 grid(N / 64, M / 256);
    gemm_bf16_kernel<<<grid, 256, 0, stream>>>(A, lda, W, ldw, C, ldc, K);
  };

  // 2) q latent: tmpA = x @ w_q_down^T ; rmsnorm -> q_c
  gemm(xb, D_MODEL, wqd, D_MODEL, tmpA, DCQ, BS, DCQ, D_MODEL);
  rmsnorm_kernel<<<BS, 256, 0, stream>>>(tmpA, DCQ, DCQ, q_norm_w, q_c, DCQ);

  // 3) kv latent: tmpB = x @ w_kv_down^T ; rmsnorm(512) -> ckv_c ; tail -> krope
  gemm(xb, D_MODEL, wkvd, D_MODEL, tmpB, 576, BS, 576, D_MODEL);
  rmsnorm_kernel<<<BS, 256, 0, stream>>>(tmpB, 576, DCKV, kv_norm_w, ckv_c, DCKV);
  extract_krope_kernel<<<(BS * DROPE + 255) / 256, 256, 0, stream>>>(tmpB, krope);

  // 4) q up-projection + RoPE -> query
  gemm(q_c, DCQ, wqu, DCQ, tmpA, H_NUM * QHD, BS, H_NUM * QHD, DCQ);
  rope_q_kernel<<<BS, 256, 0, stream>>>(tmpA, pos_ids, query);

  // 5) kv up-projection + split/RoPE -> key, vT
  gemm(ckv_c, DCKV, wkvu, DCKV, tmpB, H_NUM * 256, BS, H_NUM * 256, DCKV);
  kv_split_kernel<<<BS, 256, 0, stream>>>(tmpB, krope, pos_ids, keyb, vT);

  // 6) causal flash attention -> attn (bf16, BS x 2048)
  {
    dim3 grid(S_LEN / 128, H_NUM, B_SZ);
    flash_attn_kernel<<<grid, 256, 0, stream>>>(query, keyb, vT, attn);
  }

  // 7) output projection: out = attn @ w_out^T (fp32 to d_out)
  gemm(attn, H_NUM * DV, woutb, H_NUM * DV, out, D_MODEL, BS, D_MODEL, H_NUM * DV);
}
